// MultiHeadAttention_26225070309648
// MI455X (gfx1250) — matrix operand order repaired and checked by reading, not compiled, not run on hardware
//
#include <hip/hip_runtime.h>

// ---------------------------------------------------------------------------
// MI455X (gfx1250) wave32 WMMA implementation of windowed-conv attention.
//  - score-axis window conv folded into Q (linearity) -> flash attention
//  - all matmuls: v_wmma_f32_16x16x32_bf16 (bf16 in, f32 accumulate)
//  - all global->LDS tile staging: global_load_async_to_lds_b128 (ASYNCcnt),
//    double-buffered so the DMA of tile i+1 overlaps the WMMAs of tile i
//  - attention computes S^T = K * Q^T so K is consumed row-major and the
//    P matrix converts C-layout -> A-layout entirely in registers
// ---------------------------------------------------------------------------

#define DIMC   512
#define HEADS  8
#define DHEAD  64
#define BATCH  8
#define SEQ    1024
#define MROWS  (BATCH * SEQ)   // 8192

typedef __attribute__((ext_vector_type(16))) __bf16 bf16x16;
typedef __attribute__((ext_vector_type(8)))  float  f32x8;

__device__ __forceinline__ unsigned short f2bf(float f) {
  union { float f; unsigned u; } v; v.f = f;
  unsigned r = v.u + 0x7fffu + ((v.u >> 16) & 1u);   // round-to-nearest-even
  return (unsigned short)(r >> 16);
}

union FragB {
  bf16x16 v;
  uint4 q[2];
  unsigned short u[16];
};

__device__ __forceinline__ f32x8 wmma_bf16(bf16x16 a, bf16x16 b, f32x8 c) {
  // 8 args: (neg_a, A, neg_b, B, c_mod, C, reuse_a, reuse_b)
  return __builtin_amdgcn_wmma_f32_16x16x32_bf16(false, a, false, b, (short)0, c,
                                                 false, false);
}

// Flat LDS pointer -> 32-bit LDS byte offset (addr[31:0] per ISA aperture map)
__device__ __forceinline__ unsigned lds_off(const void* p) {
  return (unsigned)(unsigned long long)p;
}

// CDNA5 async DMA: 16B global -> LDS, tracked with ASYNCcnt (no VGPR staging)
__device__ __forceinline__ void async_ld_b128(unsigned ldsAddr, const void* gaddr) {
  asm volatile("global_load_async_to_lds_b128 %0, %1, off"
               :: "v"(ldsAddr), "v"(gaddr) : "memory");
}
__device__ __forceinline__ void wait_async0() {
  asm volatile("s_wait_asynccnt 0" ::: "memory");
}

// ---------------------------------------------------------------------------
// f32 -> bf16 conversion
// ---------------------------------------------------------------------------
__global__ __launch_bounds__(256)
void cvt_f32_bf16(const float* __restrict__ in, unsigned short* __restrict__ out, int n) {
  int i = blockIdx.x * blockDim.x + threadIdx.x;
  if (i < n) out[i] = f2bf(in[i]);
}

// ---------------------------------------------------------------------------
// GEMM: C[M,N] = A[M,K](bf16) * B[K,N](bf16) + bias
// Block tile 128x64, 4 waves (2x2), each wave 64x32 via 4x2 WMMA frags.
// Double-buffered LDS fed by async-to-LDS DMA.
// ---------------------------------------------------------------------------
#define GTM 128
#define GTN 64
#define GKB 32

__global__ __launch_bounds__(128)
void gemm_bf16(const unsigned short* __restrict__ A,
               const unsigned short* __restrict__ B,
               const float* __restrict__ bias,
               float* __restrict__ Cf,
               unsigned short* __restrict__ Cb,
               int M, int N, int K)
{
  __shared__ unsigned short As[2][GTM][GKB + 8];   // stride 40 (80B, 16B-aligned)
  __shared__ unsigned short Bs[2][GKB][GTN + 8];   // stride 72 (144B, 16B-aligned)

  const int tid  = threadIdx.x;
  const int lane = tid & 31;
  const int wave = tid >> 5;
  const int wr   = wave >> 1;
  const int wc   = wave & 1;
  const int m0   = blockIdx.y * GTM;
  const int n0   = blockIdx.x * GTN;
  const int half = lane >> 4;
  const int l15  = lane & 15;

  f32x8 acc[4][2];
#pragma unroll
  for (int i = 0; i < 4; ++i)
#pragma unroll
    for (int j = 0; j < 2; ++j)
#pragma unroll
      for (int r = 0; r < 8; ++r) acc[i][j][r] = 0.f;

  // async DMA of one (A,B) K-slice into buffer `buf`
  auto load_tiles = [&](int kb, int buf) {
#pragma unroll
    for (int i = 0; i < 4; ++i) {              // A: 128x32 = 512 x 16B chunks
      int idx = tid + i * 128;
      int row = idx >> 2, seg = idx & 3;
      async_ld_b128(lds_off(&As[buf][row][seg * 8]),
                    &A[(size_t)(m0 + row) * K + kb + seg * 8]);
    }
#pragma unroll
    for (int i = 0; i < 2; ++i) {              // B: 32x64 = 256 x 16B chunks
      int idx = tid + i * 128;
      int row = idx >> 3, seg = idx & 7;
      async_ld_b128(lds_off(&Bs[buf][row][seg * 8]),
                    &B[(size_t)(kb + row) * N + n0 + seg * 8]);
    }
  };

  load_tiles(0, 0);
  wait_async0();
  __syncthreads();

  int buf = 0;
  for (int kb = 0; kb < K; kb += GKB) {
    if (kb + GKB < K) load_tiles(kb + GKB, buf ^ 1);   // DMA next tile now

    FragB aF[4], bF[2];
#pragma unroll
    for (int fm = 0; fm < 4; ++fm) {           // A frag: row=lane&15, K-chunks
      int row = wr * 64 + fm * 16 + l15;
      aF[fm].q[0] = *(const uint4*)&As[buf][row][half * 8];
      aF[fm].q[1] = *(const uint4*)&As[buf][row][16 + half * 8];
    }
#pragma unroll
    for (int fn = 0; fn < 2; ++fn) {           // B frag: N-col=lane&15, elems=K-chunks
      int c0 = wc * 32 + fn * 16;
#pragma unroll
      for (int s = 0; s < 16; ++s)
        bF[fn].u[s] = Bs[buf][(s & 8) * 2 + half * 8 + (s & 7)][c0 + l15];
    }
#pragma unroll
    for (int fm = 0; fm < 4; ++fm)
#pragma unroll
      for (int fn = 0; fn < 2; ++fn)
        acc[fm][fn] = wmma_bf16(aF[fm].v, bF[fn].v, acc[fm][fn]);

    wait_async0();                             // next tile landed in LDS
    __syncthreads();
    buf ^= 1;
  }

  // C/D layout: VGPR r -> row = r + 8*half, col = lane&15
#pragma unroll
  for (int fm = 0; fm < 4; ++fm) {
    int rowBase = m0 + wr * 64 + fm * 16 + half * 8;
#pragma unroll
    for (int fn = 0; fn < 2; ++fn) {
      int col = n0 + wc * 32 + fn * 16 + l15;
      float bv = bias[col];
#pragma unroll
      for (int r = 0; r < 8; ++r) {
        float val = acc[fm][fn][r] + bv;
        size_t off = (size_t)(rowBase + r) * N + col;
        if (Cf) Cf[off] = val;
        if (Cb) Cb[off] = f2bf(val);
      }
    }
  }
}

// ---------------------------------------------------------------------------
// Fold score-axis window conv into Q: Qt[b,t,c] = 0.125 * sum_d 0.7^|d| Q[b,t+d,c]
// ---------------------------------------------------------------------------
__global__ __launch_bounds__(256)
void conv_q(const float* __restrict__ Q, unsigned short* __restrict__ Qt) {
  int gid = blockIdx.x * blockDim.x + threadIdx.x;   // over MROWS * DIMC/4
  int c0 = (gid & (DIMC / 4 - 1)) * 4;
  int bt = gid / (DIMC / 4);
  int t  = bt & (SEQ - 1);
  int b  = bt / SEQ;

  float w15 = 1.f;
#pragma unroll
  for (int i = 0; i < 15; ++i) w15 *= 0.7f;          // 0.7^15

  float a0 = 0.f, a1 = 0.f, a2 = 0.f, a3 = 0.f;
  float wd = w15;
#pragma unroll
  for (int d = -15; d <= 15; ++d) {
    int tt = t + d;
    if (tt >= 0 && tt < SEQ) {
      const float4 qv = *(const float4*)&Q[((size_t)b * SEQ + tt) * DIMC + c0];
      a0 += wd * qv.x; a1 += wd * qv.y; a2 += wd * qv.z; a3 += wd * qv.w;
    }
    wd = (d < 0) ? wd * (1.0f / 0.7f) : wd * 0.7f;
  }
  const float sc = 0.125f;                           // 1/sqrt(DHEAD)
  size_t o = ((size_t)b * SEQ + t) * DIMC + c0;
  Qt[o + 0] = f2bf(a0 * sc); Qt[o + 1] = f2bf(a1 * sc);
  Qt[o + 2] = f2bf(a2 * sc); Qt[o + 3] = f2bf(a3 * sc);
}

// ---------------------------------------------------------------------------
// Flash attention, one (b,h, 64-query tile) per 128-thread block (4 waves),
// each wave owns 16 queries. Computes S^T = K * Qt^T so:
//   - K/V tiles stream row-major through double-buffered LDS via async DMA
//   - Qt is transposed once into LDS; its B-frags are loop-invariant
//   - P (A-layout) is an in-register permutation of S^T (C-layout)
// Per-lane softmax state: query = lane&15 (mirrored across halves).
// ---------------------------------------------------------------------------
__global__ __launch_bounds__(128)
void attn_wmma(const unsigned short* __restrict__ Qt,
               const unsigned short* __restrict__ Kb,
               const unsigned short* __restrict__ Vb,
               unsigned short* __restrict__ Ob)
{
  __shared__ unsigned short QsT[DHEAD][64 + 8];     // dhead x queries
  __shared__ unsigned short Kt[2][32][DHEAD + 8];   // keys x dhead
  __shared__ unsigned short Vt[2][32][DHEAD + 8];   // keys x dhead

  const int tid  = threadIdx.x;
  const int lane = tid & 31;
  const int wave = tid >> 5;
  const int half = lane >> 4;
  const int l15  = lane & 15;
  const int bh = blockIdx.y, b = bh >> 3, h = bh & 7;
  const int q0 = blockIdx.x * 64;

  auto load_kv = [&](int kb, int buf) {
    const size_t baseK = ((size_t)b * SEQ + kb) * DIMC + h * DHEAD;
#pragma unroll
    for (int i = 0; i < 2; ++i) {               // 32x64 bf16 = 256 x 16B each
      int idx = tid + i * 128;
      int key = idx >> 3, seg = idx & 7;
      async_ld_b128(lds_off(&Kt[buf][key][seg * 8]),
                    &Kb[baseK + (size_t)key * DIMC + seg * 8]);
      async_ld_b128(lds_off(&Vt[buf][key][seg * 8]),
                    &Vb[baseK + (size_t)key * DIMC + seg * 8]);
    }
  };

  load_kv(0, 0);                                // first K/V tile in flight

  // transpose 64x64 Q tile into LDS (once; coalesced global reads)
  const size_t baseQ = ((size_t)b * SEQ + q0) * DIMC + h * DHEAD;
#pragma unroll
  for (int i = 0; i < 32; ++i) {
    int idx = tid + i * 128;                    // 0..4095
    int q = idx >> 6, d = idx & 63;
    QsT[d][q] = Qt[baseQ + (size_t)q * DIMC + d];
  }
  wait_async0();
  __syncthreads();

  // Loop-invariant B-frags of Qt^T: lane&15 = query, elems = d-chunks
  FragB bQ[2];
#pragma unroll
  for (int kd = 0; kd < 2; ++kd) {
#pragma unroll
    for (int s = 0; s < 16; ++s)
      bQ[kd].u[s] = QsT[kd * 32 + (s & 8) * 2 + half * 8 + (s & 7)][wave * 16 + l15];
  }

  f32x8 oAcc[4];
#pragma unroll
  for (int f = 0; f < 4; ++f)
#pragma unroll
    for (int r = 0; r < 8; ++r) oAcc[f][r] = 0.f;
  float m = -1e30f, l = 0.f;                    // softmax state for query l15

  int buf = 0;
  for (int kb = 0; kb < SEQ; kb += 32) {
    if (kb + 32 < SEQ) load_kv(kb + 32, buf ^ 1);   // DMA next tile now

    // S^T (32 keys x 16 queries): A = K row-major, B = Qt^T
    f32x8 sT[2];
#pragma unroll
    for (int kf = 0; kf < 2; ++kf)
#pragma unroll
      for (int r = 0; r < 8; ++r) sT[kf][r] = 0.f;
#pragma unroll
    for (int kf = 0; kf < 2; ++kf) {
#pragma unroll
      for (int kd = 0; kd < 2; ++kd) {
        FragB aK;                               // row = key, elems = d-chunk
        int row = kf * 16 + l15;
        aK.q[0] = *(const uint4*)&Kt[buf][row][kd * 32 + half * 8];
        aK.q[1] = *(const uint4*)&Kt[buf][row][kd * 32 + 16 + half * 8];
        sT[kf] = wmma_bf16(aK.v, bQ[kd].v, sT[kf]);
      }
    }

    // Online softmax: lane holds 16 of the 32 keys for query l15
    float smax = -1e30f;
#pragma unroll
    for (int g = 0; g < 2; ++g)
#pragma unroll
      for (int r = 0; r < 8; ++r) smax = fmaxf(smax, sT[g][r]);
    smax = fmaxf(smax, __shfl_xor(smax, 16, 32));
    float mnew = fmaxf(m, smax);
    float esc  = __expf(m - mnew);

    // P in A-layout directly: elem e=(g*8+w) -> key g*16 + 8*half + w == sT[g][w]
    FragB pF;
    float psum = 0.f;
#pragma unroll
    for (int g = 0; g < 2; ++g)
#pragma unroll
      for (int w = 0; w < 8; ++w) {
        float p = __expf(sT[g][w] - mnew);
        psum += p;
        pF.u[g * 8 + w] = f2bf(p);
      }
    psum += __shfl_xor(psum, 16, 32);
    l = l * esc + psum;
    m = mnew;

    // Rescale O: row r of oAcc is query (r + 8*half); esc lives in that lane
#pragma unroll
    for (int r = 0; r < 8; ++r) {
      float e = __shfl(esc, r + 8 * half, 32);
#pragma unroll
      for (int df = 0; df < 4; ++df) oAcc[df][r] *= e;
    }

    // O += P * V
#pragma unroll
    for (int df = 0; df < 4; ++df) {
      FragB vF;                                 // col = d = lane&15, elems = keys
#pragma unroll
      for (int s = 0; s < 16; ++s)
        vF.u[s] = Vt[buf][(s & 8) * 2 + half * 8 + (s & 7)][df * 16 + l15];
      oAcc[df] = wmma_bf16(pF.v, vF.v, oAcc[df]);
    }

    wait_async0();
    __syncthreads();
    buf ^= 1;
  }

  // Normalize and store bf16 in [B,T,H*D] layout
#pragma unroll
  for (int r = 0; r < 8; ++r) {
    float linv = 1.f / __shfl(l, r + 8 * half, 32);
    int q = q0 + wave * 16 + half * 8 + r;
    size_t base = ((size_t)b * SEQ + q) * DIMC + h * DHEAD;
#pragma unroll
    for (int df = 0; df < 4; ++df)
      Ob[base + df * 16 + l15] = f2bf(oAcc[df][r] * linv);
  }
}

// ---------------------------------------------------------------------------
// Launch
// ---------------------------------------------------------------------------
extern "C" void kernel_launch(void* const* d_in, const int* in_sizes, int n_in,
                              void* d_out, int out_size, void* d_ws, size_t ws_size,
                              hipStream_t stream) {
  const float* x  = (const float*)d_in[0];
  const float* Wq = (const float*)d_in[1];
  const float* bq = (const float*)d_in[2];
  const float* Wk = (const float*)d_in[3];
  const float* bk = (const float*)d_in[4];
  const float* Wv = (const float*)d_in[5];
  const float* bv = (const float*)d_in[6];
  const float* Wo = (const float*)d_in[7];
  const float* bo = (const float*)d_in[8];
  float* out = (float*)d_out;

  const size_t NX = (size_t)MROWS * DIMC;      // 4M elements
  const size_t NW = (size_t)DIMC * DIMC;       // 256K elements

  // Workspace layout (~58 MiB, all regions 16B aligned)
  unsigned short* xbf = (unsigned short*)d_ws;
  unsigned short* Wqb = xbf + NX;
  unsigned short* Wkb = Wqb + NW;
  unsigned short* Wvb = Wkb + NW;
  unsigned short* Wob = Wvb + NW;
  float*          Qf  = (float*)(Wob + NW);
  unsigned short* Kbf = (unsigned short*)(Qf + NX);
  unsigned short* Vbf = Kbf + NX;
  unsigned short* Qtb = Vbf + NX;
  unsigned short* Abf = Qtb + NX;

  // 1) bf16 conversions
  cvt_f32_bf16<<<(int)((NX + 255) / 256), 256, 0, stream>>>(x,  xbf, (int)NX);
  cvt_f32_bf16<<<(int)((NW + 255) / 256), 256, 0, stream>>>(Wq, Wqb, (int)NW);
  cvt_f32_bf16<<<(int)((NW + 255) / 256), 256, 0, stream>>>(Wk, Wkb, (int)NW);
  cvt_f32_bf16<<<(int)((NW + 255) / 256), 256, 0, stream>>>(Wv, Wvb, (int)NW);
  cvt_f32_bf16<<<(int)((NW + 255) / 256), 256, 0, stream>>>(Wo, Wob, (int)NW);

  // 2) Q/K/V projections (Q kept f32 for the window conv)
  dim3 gg(DIMC / GTN, MROWS / GTM);            // (8, 64)
  gemm_bf16<<<gg, 128, 0, stream>>>(xbf, Wqb, bq, Qf,      nullptr, MROWS, DIMC, DIMC);
  gemm_bf16<<<gg, 128, 0, stream>>>(xbf, Wkb, bk, nullptr, Kbf,     MROWS, DIMC, DIMC);
  gemm_bf16<<<gg, 128, 0, stream>>>(xbf, Wvb, bv, nullptr, Vbf,     MROWS, DIMC, DIMC);

  // 3) fold score-axis conv + 1/sqrt(dhead) into Q
  conv_q<<<(int)(NX / 4 / 256), 256, 0, stream>>>(Qf, Qtb);

  // 4) flash attention
  attn_wmma<<<dim3(SEQ / 64, BATCH * HEADS), 128, 0, stream>>>(Qtb, Kbf, Vbf, Abf);

  // 5) output projection -> f32 d_out
  gemm_bf16<<<gg, 128, 0, stream>>>(Abf, Wob, bo, out, nullptr, MROWS, DIMC, DIMC);
}
